// NodeGATLayer_91216515432630
// MI455X (gfx1250) — compile-verified, matrix operand order checked by reading, not run on hardware
//
#include <hip/hip_runtime.h>
#include <hip/hip_bf16.h>

#define DFEAT 128

typedef __attribute__((ext_vector_type(16))) _Float16 v16h;
typedef __attribute__((ext_vector_type(8)))  _Float16 v8h;
typedef __attribute__((ext_vector_type(8)))  float    v8f;

// Order-preserving int encoding of float (involution: encode == decode).
__device__ __forceinline__ int ord_xform(int i) {
    return i >= 0 ? i : (i ^ 0x7fffffff);
}
__device__ __forceinline__ int   ord_encode(float f) { return ord_xform(__float_as_int(f)); }
__device__ __forceinline__ float ord_decode(int i)   { return __int_as_float(ord_xform(i)); }

// ---------------------------------------------------------------------------
// K0: zero output, init per-node segment-max (ordered-int -inf) and denom.
// ---------------------------------------------------------------------------
__global__ void gat_init_kernel(float* __restrict__ out, int out_n,
                                int* __restrict__ segmax_i,
                                float* __restrict__ denom, int n_nodes) {
    int t = blockIdx.x * blockDim.x + threadIdx.x;
    if (t < out_n) out[t] = 0.0f;
    if (t < n_nodes) { segmax_i[t] = (int)0x80000000; denom[t] = 0.0f; }
}

// ---------------------------------------------------------------------------
// K1: logit[i] = tanh(N @ W1^T)[i,:] . w2  via v_wmma_f32_16x16x32_f16.
// Block = 256 threads = 8 waves; each wave owns a 16-row tile; block = 128 rows.
// W1 staged f32->f16 into LDS with 8-half row padding (row stride 272B) so the
// 16 lanes of a B-fragment read hit 64 distinct banks.
// ---------------------------------------------------------------------------
__global__ __launch_bounds__(256) void gat_logit_wmma_kernel(
    const float* __restrict__ Nmat, const float* __restrict__ W1,
    const float* __restrict__ w2, float* __restrict__ logit, int n_nodes) {
    __shared__ __align__(16) _Float16 sW1[128 * 136];

    const int tid = threadIdx.x;
    // Stage W1 (row-major [j][k], which is exactly B^T layout we need).
    for (int idx = tid; idx < 128 * 128; idx += 256) {
        int r = idx >> 7, c = idx & 127;
        sW1[r * 136 + c] = (_Float16)W1[idx];
    }
    __syncthreads();

    const int wave = tid >> 5;
    const int lane = tid & 31;
    const int lm   = lane & 15;   // M (for A) / N (for B) index within tile
    const int hf   = lane >> 4;   // lane half selects K sub-block
    const int rowBase = blockIdx.x * 128 + wave * 16;
    const int row     = rowBase + lm;
    const bool rowOK  = row < n_nodes;

    v8f acc[8];
#pragma unroll
    for (int jt = 0; jt < 8; ++jt) acc[jt] = (v8f){0.f,0.f,0.f,0.f,0.f,0.f,0.f,0.f};

    const float4* rowp = reinterpret_cast<const float4*>(Nmat + (size_t)row * DFEAT);

#pragma unroll
    for (int kt = 0; kt < 128; kt += 32) {
        // A fragment: 16x32 f16, lane holds row `lm`, K groups per ISA layout:
        // regs 0..3 -> K = kt + hf*8 + 0..7 ; regs 4..7 -> K = kt + 16 + hf*8 + 0..7
        v16h a;
        if (rowOK) {
            const int b0 = (kt + hf * 8) >> 2;         // float4 index of first chunk
            float4 c0 = rowp[b0];
            float4 c1 = rowp[b0 + 1];
            float4 c2 = rowp[b0 + 4];                  // kt + 16 + hf*8
            float4 c3 = rowp[b0 + 5];
            a[0]=(_Float16)c0.x; a[1]=(_Float16)c0.y; a[2]=(_Float16)c0.z; a[3]=(_Float16)c0.w;
            a[4]=(_Float16)c1.x; a[5]=(_Float16)c1.y; a[6]=(_Float16)c1.z; a[7]=(_Float16)c1.w;
            a[8]=(_Float16)c2.x; a[9]=(_Float16)c2.y; a[10]=(_Float16)c2.z; a[11]=(_Float16)c2.w;
            a[12]=(_Float16)c3.x; a[13]=(_Float16)c3.y; a[14]=(_Float16)c3.z; a[15]=(_Float16)c3.w;
        } else {
#pragma unroll
            for (int i = 0; i < 16; ++i) a[i] = (_Float16)0.0f;
        }

#pragma unroll
        for (int jt = 0; jt < 8; ++jt) {
            // B fragment: B[k][n] = W1[j][k] with j = jt*16 + lm; two 16B LDS reads.
            const int j = jt * 16 + lm;
            const v8h lo = *reinterpret_cast<const v8h*>(&sW1[j * 136 + kt + hf * 8]);
            const v8h hi = *reinterpret_cast<const v8h*>(&sW1[j * 136 + kt + 16 + hf * 8]);
            v16h b;
#pragma unroll
            for (int i = 0; i < 8; ++i) { b[i] = lo[i]; b[8 + i] = hi[i]; }
            acc[jt] = __builtin_amdgcn_wmma_f32_16x16x32_f16(
                /*neg_a=*/false, a, /*neg_b=*/false, b,
                /*c_mod=*/(short)0, acc[jt], /*reuse_a=*/false, /*reuse_b=*/false);
        }
    }

    // Epilogue: logit[m] = sum_j tanh(T[m][j]) * w2[j].
    // C layout: lane holds column j = jt*16 + lm; reg r holds row m = r + 8*hf.
    float w2v[8];
#pragma unroll
    for (int jt = 0; jt < 8; ++jt) w2v[jt] = w2[jt * 16 + lm];

#pragma unroll
    for (int r = 0; r < 8; ++r) {
        float s = 0.0f;
#pragma unroll
        for (int jt = 0; jt < 8; ++jt) s += tanhf(acc[jt][r]) * w2v[jt];
        // Reduce across the 16 lanes of each half (masks never cross bit 4).
        s += __shfl_xor(s, 1);
        s += __shfl_xor(s, 2);
        s += __shfl_xor(s, 4);
        s += __shfl_xor(s, 8);
        const int grow = rowBase + hf * 8 + r;
        if (lm == 0 && grow < n_nodes) logit[grow] = s;
    }
}

// ---------------------------------------------------------------------------
// K2: per-edge segment max via atomicMax on order-encoded ints.
// ---------------------------------------------------------------------------
__global__ void gat_edge_max_kernel(const int* __restrict__ src, const int* __restrict__ dst,
                                    const float* __restrict__ logit,
                                    int* __restrict__ segmax_i, int n_edges) {
    int t = blockIdx.x * blockDim.x + threadIdx.x;
    if (t >= n_edges) return;
    atomicMax(&segmax_i[dst[t]], ord_encode(logit[src[t]]));
}

// ---------------------------------------------------------------------------
// K3: per-edge exp and segment-sum of denominators.
// ---------------------------------------------------------------------------
__global__ void gat_edge_denom_kernel(const int* __restrict__ src, const int* __restrict__ dst,
                                      const float* __restrict__ logit,
                                      const int* __restrict__ segmax_i,
                                      float* __restrict__ denom, int n_edges) {
    int t = blockIdx.x * blockDim.x + threadIdx.x;
    if (t >= n_edges) return;
    const int d = dst[t];
    const float ex = __expf(logit[src[t]] - ord_decode(segmax_i[d]));
    atomicAdd(&denom[d], ex);
}

// ---------------------------------------------------------------------------
// K4: one wave per edge: out[dst] += a * N[src]. Lane l handles floats 4l..4l+3.
// Scalar weight recomputed per-lane (uniform broadcast load, L2 resident).
// ---------------------------------------------------------------------------
__global__ __launch_bounds__(256) void gat_scatter_kernel(
    const float* __restrict__ Nmat, const int* __restrict__ src, const int* __restrict__ dst,
    const float* __restrict__ logit, const int* __restrict__ segmax_i,
    const float* __restrict__ denom, float* __restrict__ out, int n_edges) {
    const int wid  = blockIdx.x * (blockDim.x >> 5) + (threadIdx.x >> 5);
    const int lane = threadIdx.x & 31;
    if (wid >= n_edges) return;
    const int s = src[wid];
    const int d = dst[wid];
    const float a = __expf(logit[s] - ord_decode(segmax_i[d])) / denom[d];
    const float4 v = reinterpret_cast<const float4*>(Nmat)[(size_t)s * 32 + lane];
    float* o = out + (size_t)d * DFEAT + lane * 4;
    atomicAdd(o + 0, a * v.x);
    atomicAdd(o + 1, a * v.y);
    atomicAdd(o + 2, a * v.z);
    atomicAdd(o + 3, a * v.w);
}

// ---------------------------------------------------------------------------
extern "C" void kernel_launch(void* const* d_in, const int* in_sizes, int n_in,
                              void* d_out, int out_size, void* d_ws, size_t ws_size,
                              hipStream_t stream) {
    const float* Nmat = (const float*)d_in[0];
    const float* W1   = (const float*)d_in[1];
    const float* w2   = (const float*)d_in[2];
    const int*   src  = (const int*)d_in[3];
    const int*   dst  = (const int*)d_in[4];
    const int n_nodes = in_sizes[0] / DFEAT;
    const int n_edges = in_sizes[3];
    float* out = (float*)d_out;

    // Workspace layout: logit[f32 n_nodes] | segmax_i[i32 n_nodes] | denom[f32 n_nodes]
    float* logit    = (float*)d_ws;
    int*   segmax_i = (int*)((char*)d_ws + (size_t)n_nodes * 4);
    float* denom    = (float*)((char*)d_ws + (size_t)n_nodes * 8);

    const int initN = out_size > n_nodes ? out_size : n_nodes;
    gat_init_kernel<<<(initN + 255) / 256, 256, 0, stream>>>(out, out_size, segmax_i, denom, n_nodes);

    gat_logit_wmma_kernel<<<(n_nodes + 127) / 128, 256, 0, stream>>>(Nmat, W1, w2, logit, n_nodes);

    const int eb = (n_edges + 255) / 256;
    gat_edge_max_kernel<<<eb, 256, 0, stream>>>(src, dst, logit, segmax_i, n_edges);
    gat_edge_denom_kernel<<<eb, 256, 0, stream>>>(src, dst, logit, segmax_i, denom, n_edges);

    gat_scatter_kernel<<<(n_edges + 7) / 8, 256, 0, stream>>>(Nmat, src, dst, logit, segmax_i,
                                                              denom, out, n_edges);
}
